// NeuralODE_8890582303280
// MI455X (gfx1250) — compile-verified, matrix operand order checked by reading, not run on hardware
//
#include <hip/hip_runtime.h>
#include <hip/hip_bf16.h>

typedef __attribute__((ext_vector_type(16))) _Float16 v16h;
typedef __attribute__((ext_vector_type(8)))  _Float16 v8h;
typedef __attribute__((ext_vector_type(4)))  _Float16 v4h;
typedef __attribute__((ext_vector_type(8)))  float    v8f;
typedef __attribute__((ext_vector_type(4)))  float    v4f;

#define BATCH  1024
#define DIM    64
#define HID    256
#define TM     16          // rows per workgroup (WMMA M)
#define NW     8           // waves per workgroup
#define NTHR   (NW * 32)

__device__ __forceinline__ float tanh_fast(float x) {
#if __has_builtin(__builtin_amdgcn_tanhf)
    return __builtin_amdgcn_tanhf(x);        // v_tanh_f32 (gfx1250 TRANS op)
#else
    float ax = fabsf(x);
    float e  = __expf(2.0f * ax);
    float r  = 1.0f - 2.0f * __builtin_amdgcn_rcpf(e + 1.0f);  // v_rcp_f32
    return copysignf(r, x);
#endif
}

__device__ __forceinline__ v8f wmma_f16(v16h a, v16h b, v8f c) {
    // D = A(16x32 f16) * B(32x16 f16) + C(16x16 f32)
    return __builtin_amdgcn_wmma_f32_16x16x32_f16(
        /*neg_a=*/false, a, /*neg_b=*/false, b,
        /*c_mod=*/(short)0, c, /*reuse_a=*/false, /*reuse_b=*/false);
}

// Build a 16x32 f16 A-fragment straight from an f16 row-major LDS tile.
// Lane L<16: row=L, chunks [kb..kb+7],[kb+16..kb+23]; lane>=16: kb += 8.
__device__ __forceinline__ v16h build_a_f16(const _Float16* src, int rstride,
                                            int row, int kbase) {
    const v8h lo = *(const v8h*)&src[row * rstride + kbase];
    const v8h hh = *(const v8h*)&src[row * rstride + kbase + 16];
    v16h a;
#pragma unroll
    for (int i = 0; i < 8; ++i) { a[i] = lo[i]; a[8 + i] = hh[i]; }
    return a;
}

__global__ __launch_bounds__(NTHR)
void node_rk4_kernel(const float* __restrict__ y0, const float* __restrict__ tarr,
                     const float* __restrict__ W1, const float* __restrict__ b1,
                     const float* __restrict__ W2, const float* __restrict__ b2,
                     float* __restrict__ out, int nsteps) {
    __shared__ __align__(16) float    yS [TM * DIM];   // current state (f32)
    __shared__ __align__(16) float    kS [TM * DIM];   // func output (k_i)
    __shared__ __align__(16) float    acS[TM * DIM];   // RK4 accumulator
    __shared__ __align__(16) _Float16 yhS[TM * DIM];   // f16 mirror of func input
    __shared__ __align__(16) _Float16 hS [TM * HID];   // tanh hidden (f16)

    const int tid  = threadIdx.x;
    const int w    = tid >> 5;           // wave id 0..7
    const int lane = tid & 31;
    const int ln   = lane & 15;
    const int hi   = lane >> 4;          // half-wave select
    const int mb   = 8 * hi;             // C/D row base for this lane
    const int wg   = blockIdx.x;
    const int e4   = 4 * tid;            // this thread's 4 contiguous elements

    // ---- Preload resident f16 B-fragments (reused 999*4 times) ----
    v16h B1f[2][2];                      // GEMM1: cols [32w,32w+32), K=64 in 2 steps
#pragma unroll
    for (int j = 0; j < 2; ++j) {
#pragma unroll
        for (int s = 0; s < 2; ++s) {
            const int n   = 32 * w + 16 * j + ln;
            const int kst = 32 * s + 8 * hi;
            v16h b;
#pragma unroll
            for (int i = 0; i < 8; ++i) b[i]     = (_Float16)W1[(kst + i) * HID + n];
#pragma unroll
            for (int i = 0; i < 8; ++i) b[8 + i] = (_Float16)W1[(kst + 16 + i) * HID + n];
            B1f[j][s] = b;
        }
    }
    v16h B2f[4];                         // GEMM2: K-slice [32w,32w+32), 4 N-tiles
#pragma unroll
    for (int tt = 0; tt < 4; ++tt) {
        const int n   = 16 * tt + ln;
        const int kst = 32 * w + 8 * hi;
        v16h b;
#pragma unroll
        for (int i = 0; i < 8; ++i) b[i]     = (_Float16)W2[(kst + i) * DIM + n];
#pragma unroll
        for (int i = 0; i < 8; ++i) b[8 + i] = (_Float16)W2[(kst + 16 + i) * DIM + n];
        B2f[tt] = b;
    }
    const float b1v0 = b1[32 * w + ln];
    const float b1v1 = b1[32 * w + 16 + ln];
    const v4f   b2v  = *(const v4f*)&b2[e4 & (DIM - 1)];   // bias for my 4 elems

    // ---- Load y tile; emit step-0 output; seed f16 mirror and kS=b2 ----
    {
        v4f y = *(const v4f*)&y0[wg * (TM * DIM) + e4];
        *(v4f*)&yS[e4] = y;
        *(v4f*)&out[wg * (TM * DIM) + e4] = y;
        v4h yh;
#pragma unroll
        for (int j = 0; j < 4; ++j) yh[j] = (_Float16)y[j];
        *(v4h*)&yhS[e4] = yh;
        *(v4f*)&kS[e4]  = b2v;           // atomics land on the bias
    }
    __syncthreads();

    // func: kS += tanh(yhS@W1+b1)@W2   (kS pre-seeded with b2 by elementwise phase)
    auto func_eval = [&]() {
        // A-fragments of f16 state (K=64 -> two 16x32 frags), pure ds_load_b128
        const v16h A0 = build_a_f16(yhS, DIM, ln, 0  + 8 * hi);
        const v16h A1 = build_a_f16(yhS, DIM, ln, 32 + 8 * hi);

        // GEMM1 + bias + tanh -> hS (f16 row-major 16x256)
#pragma unroll
        for (int j = 0; j < 2; ++j) {
            v8f c = {};
            c = wmma_f16(A0, B1f[j][0], c);
            c = wmma_f16(A1, B1f[j][1], c);
            const float bias = j ? b1v1 : b1v0;
            const int   n    = 32 * w + 16 * j + ln;
#pragma unroll
            for (int r = 0; r < 8; ++r)
                hS[(mb + r) * HID + n] = (_Float16)tanh_fast(c[r] + bias);
        }
        __syncthreads();

        // A-fragment of h for this wave's K-slice [32w,32w+32)
        const v16h A2 = build_a_f16(hS, HID, ln, 32 * w + 8 * hi);

        // GEMM2 partials, cross-wave K reduction via ds_add_f32
#pragma unroll
        for (int tt = 0; tt < 4; ++tt) {
            v8f c = {};
            c = wmma_f16(A2, B2f[tt], c);
            const int n = 16 * tt + ln;
#pragma unroll
            for (int r = 0; r < 8; ++r)
                atomicAdd(&kS[(mb + r) * DIM + n], c[r]);
        }
        __syncthreads();
    };

    // Elementwise RK4 helper: consume kS, reset kS=b2, produce next func input.
    // acS accumulates k1+2k2+2k3; yhS stages y+c*k with c = h/2, h/2, h.
    for (int s = 0; s < nsteps; ++s) {
        const float h     = tarr[s + 1] - tarr[s];
        const float halfh = 0.5f * h;

        func_eval();                                     // k1
        {
            v4f k = *(const v4f*)&kS[e4];
            *(v4f*)&kS[e4]  = b2v;
            *(v4f*)&acS[e4] = k;
            v4f y = *(const v4f*)&yS[e4];
            v4h yh;
#pragma unroll
            for (int j = 0; j < 4; ++j) yh[j] = (_Float16)(y[j] + halfh * k[j]);
            *(v4h*)&yhS[e4] = yh;
        }
        __syncthreads();

        func_eval();                                     // k2
        {
            v4f k = *(const v4f*)&kS[e4];
            *(v4f*)&kS[e4]  = b2v;
            v4f a = *(const v4f*)&acS[e4];
#pragma unroll
            for (int j = 0; j < 4; ++j) a[j] += 2.0f * k[j];
            *(v4f*)&acS[e4] = a;
            v4f y = *(const v4f*)&yS[e4];
            v4h yh;
#pragma unroll
            for (int j = 0; j < 4; ++j) yh[j] = (_Float16)(y[j] + halfh * k[j]);
            *(v4h*)&yhS[e4] = yh;
        }
        __syncthreads();

        func_eval();                                     // k3
        {
            v4f k = *(const v4f*)&kS[e4];
            *(v4f*)&kS[e4]  = b2v;
            v4f a = *(const v4f*)&acS[e4];
#pragma unroll
            for (int j = 0; j < 4; ++j) a[j] += 2.0f * k[j];
            *(v4f*)&acS[e4] = a;
            v4f y = *(const v4f*)&yS[e4];
            v4h yh;
#pragma unroll
            for (int j = 0; j < 4; ++j) yh[j] = (_Float16)(y[j] + h * k[j]);
            *(v4h*)&yhS[e4] = yh;
        }
        __syncthreads();

        func_eval();                                     // k4
        {
            v4f k = *(const v4f*)&kS[e4];
            *(v4f*)&kS[e4] = b2v;
            v4f a = *(const v4f*)&acS[e4];
            v4f y = *(const v4f*)&yS[e4];
            v4f yn;
            const float c6 = h * (1.0f / 6.0f);
#pragma unroll
            for (int j = 0; j < 4; ++j) yn[j] = y[j] + c6 * (a[j] + k[j]);
            *(v4f*)&yS[e4] = yn;
            v4h yh;
#pragma unroll
            for (int j = 0; j < 4; ++j) yh[j] = (_Float16)yn[j];
            *(v4h*)&yhS[e4] = yh;
            *(v4f*)&out[(size_t)(s + 1) * (BATCH * DIM) + wg * (TM * DIM) + e4] = yn;
        }
        __syncthreads();
    }
}

extern "C" void kernel_launch(void* const* d_in, const int* in_sizes, int n_in,
                              void* d_out, int out_size, void* d_ws, size_t ws_size,
                              hipStream_t stream) {
    // setup_inputs order: y0, t, W1, b1, W2, b2
    const float* y0 = (const float*)d_in[0];
    const float* t  = (const float*)d_in[1];
    const float* W1 = (const float*)d_in[2];
    const float* b1 = (const float*)d_in[3];
    const float* W2 = (const float*)d_in[4];
    const float* b2 = (const float*)d_in[5];
    float* out = (float*)d_out;

    const int batch  = in_sizes[0] / DIM;     // 1024
    const int nsteps = in_sizes[1] - 1;       // 999
    const int grid   = batch / TM;            // 64 independent trajectory tiles

    node_rk4_kernel<<<grid, NTHR, 0, stream>>>(y0, t, W1, b1, W2, b2, out, nsteps);
}
